// S3Cell_6786048328284
// MI455X (gfx1250) — matrix operand order repaired and checked by reading, not compiled, not run on hardware
//
#include <hip/hip_runtime.h>
#include <hip/hip_bf16.h>

// S3Cell fused gate-block pipeline for gfx1250 (MI455X).
//  prep:  W1/W2 fp32->bf16 once (L2-resident), xh concat+select -> bf16
//  gemm:  (B x 768) @ (768 x 2048)^T bf16 WMMA, f32 accumulate,
//         async global->LDS A-tile staging (ASYNCcnt path),
//         fused LSTM-style gate epilogue; phase 1 emits input2 as bf16
//         straight into the phase-2 activation buffer.

#define B_ROWS 16384
#define IN_DIM 256
#define H_DIM  256
#define NG     512          // per-gate width (G/4)
#define DXH_K  768          // IN + 2H
#define BM     64
#define BN     32
#define BK     32
#define KT     (DXH_K / BK) // 24
#define APAD   40           // bf16 row stride in LDS (80B, 16B-aligned)

typedef __bf16 v16bf __attribute__((ext_vector_type(16)));
typedef float  v8f   __attribute__((ext_vector_type(8)));
typedef int    b128_t __attribute__((vector_size(16)));   // builtin's B128 unit

union FragB16 { v16bf v; uint4 q[2]; };
union Acc8f   { v8f   v; float f[8]; };

#if __has_builtin(__builtin_amdgcn_global_load_async_to_lds_b128)
#define USE_ASYNC_LDS 1
#else
#define USE_ASYNC_LDS 0
#endif

__device__ __forceinline__ unsigned int pack_bf16x2(float a, float b) {
  unsigned int ua = __float_as_uint(a);
  unsigned int ub = __float_as_uint(b);
  ua += 0x7fffu + ((ua >> 16) & 1u);   // round-to-nearest-even
  ub += 0x7fffu + ((ub >> 16) & 1u);
  return (ua >> 16) | (ub & 0xffff0000u);
}

__device__ __forceinline__ unsigned short bf16_1(float a) {
  unsigned int ua = __float_as_uint(a);
  ua += 0x7fffu + ((ua >> 16) & 1u);
  return (unsigned short)(ua >> 16);
}

__device__ __forceinline__ uint4 pack8(const float4 a, const float4 b) {
  uint4 r;
  r.x = pack_bf16x2(a.x, a.y);
  r.y = pack_bf16x2(a.z, a.w);
  r.z = pack_bf16x2(b.x, b.y);
  r.w = pack_bf16x2(b.z, b.w);
  return r;
}

__device__ __forceinline__ float sigmoidf_(float x) {
  return 1.0f / (1.0f + __expf(-x));
}

// 16-byte global -> LDS stage: async DMA path on CDNA5, reg round-trip fallback.
__device__ __forceinline__ void stage16(const unsigned short* __restrict__ g,
                                        unsigned short* l) {
#if USE_ASYNC_LDS
  typedef __attribute__((address_space(1))) b128_t* gptr_t;  // global ptr (flat==AS1)
  typedef __attribute__((address_space(3))) b128_t* lptr_t;  // LDS ptr = addr[31:0]
  __builtin_amdgcn_global_load_async_to_lds_b128(
      (gptr_t)(unsigned long long)(uintptr_t)g,
      (lptr_t)(unsigned int)(uintptr_t)l,
      0, 0);
#else
  *(uint4*)l = *(const uint4*)g;
#endif
}

__device__ __forceinline__ void wait_async_all() {
#if USE_ASYNC_LDS
#if __has_builtin(__builtin_amdgcn_s_wait_asynccnt)
  __builtin_amdgcn_s_wait_asynccnt(0);
#else
  asm volatile("s_wait_asynccnt 0x0" ::: "memory");
#endif
#endif
}

// ---------------- prep kernels ----------------

__global__ __launch_bounds__(256)
void convert_f32_bf16(const float* __restrict__ src,
                      unsigned short* __restrict__ dst, int n8) {
  const int idx = blockIdx.x * 256 + threadIdx.x;
  if (idx >= n8) return;
  const float4 f0 = *(const float4*)(src + (size_t)idx * 8);
  const float4 f1 = *(const float4*)(src + (size_t)idx * 8 + 4);
  *(uint4*)(dst + (size_t)idx * 8) = pack8(f0, f1);
}

// Builds bf16 xh = [xlow | mid | hi] (each 256 wide) with flag-based source
// selection, starting at column startCol (xh2's low 256 cols come from phase 1).
__global__ __launch_bounds__(256)
void build_xh_bf16(const float* __restrict__ xlow,
                   const float* __restrict__ hprev,
                   const float* __restrict__ hdel,
                   unsigned short* __restrict__ dst, int startCol,
                   const int* __restrict__ hasDp, const int* __restrict__ hasPp) {
  const int hasD = hasDp[0];
  const int hasP = hasPp[0];
  const int segs = (DXH_K - startCol) >> 3;          // 8 cols per thread
  const int gid  = blockIdx.x * 256 + threadIdx.x;
  const int row  = gid / segs;
  if (row >= B_ROWS) return;
  const int c = startCol + (gid % segs) * 8;

  const float* src; int cb;
  if (c < IN_DIM)              { src = xlow; cb = c; }
  else if (c < IN_DIM + H_DIM) { src = (hasD || hasP) ? hprev : nullptr; cb = c - IN_DIM; }
  else                         { src = hasD ? hdel : (hasP ? hprev : nullptr); cb = c - (IN_DIM + H_DIM); }

  uint4 v;
  if (src) {
    const float* p = src + (size_t)row * 256 + cb;
    v = pack8(*(const float4*)p, *(const float4*)(p + 4));
  } else {
    v = make_uint4(0u, 0u, 0u, 0u);
  }
  *(uint4*)(dst + (size_t)row * DXH_K + c) = v;
}

// ---------------- fused GEMM + gate epilogue ----------------

__global__ __launch_bounds__(256)
void s3cell_gate_gemm(int phase,
                      const unsigned short* __restrict__ xh,   // (B,768) bf16
                      const unsigned short* __restrict__ Wb,   // (2048,768) bf16
                      const float* __restrict__ bias,          // (2048,)
                      const float* __restrict__ prevC,         // (B,512)
                      const float* __restrict__ delC,          // (B,512)
                      const float* __restrict__ rawInput,      // phase 1 only
                      const float* __restrict__ scale,         // phase 1 only
                      unsigned short* __restrict__ outLowB,    // phase 1: xh2 cols 0..255 (stride 768)
                      float* __restrict__ outLowF,             // phase 2: output_t (stride 256)
                      float* __restrict__ outHigh,             // (B,256) hState / hState2
                      float* __restrict__ outNS,               // (B,512) newState / newState2
                      const int* __restrict__ hasDp,
                      const int* __restrict__ hasPp)
{
  __shared__ __align__(16) unsigned short sA[2][BM * APAD];  // double-buffered bf16 X tile
  __shared__ float sG[4][BM][BN + 1];                        // f32 gate results

  const int tid  = threadIdx.x;
  const int wave = tid >> 5;       // 8 waves
  const int g    = wave & 3;       // gate id
  const int mh   = wave >> 2;      // row half (0: rows 0..31, 1: rows 32..63)
  const int lane = tid & 31;
  const int half = lane >> 4;
  const int l16  = lane & 15;

  const int mbase = blockIdx.x * BM;
  const int nbase = blockIdx.y * BN;

  const int hasD = hasDp[0];
  const int hasP = hasPp[0];

  // A staging: 256 threads cover 64 rows x 32 k (one 16B transfer each)
  const int srow = tid >> 2;
  const int sseg = (tid & 3) * 8;
  const unsigned short* aptr = xh + (size_t)(mbase + srow) * DXH_K + sseg;
  unsigned short* lbuf0 = &sA[0][srow * APAD + sseg];
  unsigned short* lbuf1 = &sA[1][srow * APAD + sseg];

  // B fragments straight from global bf16 weights (L2-resident):
  // lane(half,n) holds column n of the 32x16 tile, K runs [8*half .. +8) and [16+8*half .. +8)
  const unsigned short* wrow0 = Wb + (size_t)(g * NG + nbase + l16) * DXH_K + 8 * half;
  const unsigned short* wrow1 = wrow0 + (size_t)16 * DXH_K;

  Acc8f acc[2][2];
#pragma unroll
  for (int mi = 0; mi < 2; ++mi)
#pragma unroll
    for (int ni = 0; ni < 2; ++ni)
#pragma unroll
      for (int r = 0; r < 8; ++r) acc[mi][ni].f[r] = 0.0f;

  stage16(aptr, lbuf0);            // prologue: k-tile 0 -> buffer 0

  for (int kt = 0; kt < KT; ++kt) {
    const int buf = kt & 1;
    const int k0  = kt * BK;

    wait_async_all();              // this wave's stage of `buf` is in LDS
    __syncthreads();               // all waves staged `buf`; all reads of buf^1 done

    // Issue next tile's DMA into the other buffer; overlaps with compute below.
    if (kt + 1 < KT)
      stage16(aptr + (size_t)(kt + 1) * BK, buf ? lbuf0 : lbuf1);

    FragB16 a0, a1, b0, b1;
    const unsigned short* wp0 = wrow0 + k0;
    b0.q[0] = *(const uint4*)(wp0);
    b0.q[1] = *(const uint4*)(wp0 + 16);
    const unsigned short* wp1 = wrow1 + k0;
    b1.q[0] = *(const uint4*)(wp1);
    b1.q[1] = *(const uint4*)(wp1 + 16);

    // A lane(half,m): K runs [8*half .. +8) and [16+8*half .. +8)
    const int r0 = (mh * 32 + l16) * APAD + 8 * half;
    a0.q[0] = *(const uint4*)&sA[buf][r0];
    a0.q[1] = *(const uint4*)&sA[buf][r0 + 16];
    const int r1 = (mh * 32 + 16 + l16) * APAD + 8 * half;
    a1.q[0] = *(const uint4*)&sA[buf][r1];
    a1.q[1] = *(const uint4*)&sA[buf][r1 + 16];

    acc[0][0].v = __builtin_amdgcn_wmma_f32_16x16x32_bf16(false, a0.v, false, b0.v, (short)0, acc[0][0].v, false, false);
    acc[0][1].v = __builtin_amdgcn_wmma_f32_16x16x32_bf16(false, a0.v, false, b1.v, (short)0, acc[0][1].v, false, false);
    acc[1][0].v = __builtin_amdgcn_wmma_f32_16x16x32_bf16(false, a1.v, false, b0.v, (short)0, acc[1][0].v, false, false);
    acc[1][1].v = __builtin_amdgcn_wmma_f32_16x16x32_bf16(false, a1.v, false, b1.v, (short)0, acc[1][1].v, false, false);

    if (kt + 1 < KT) __builtin_prefetch(wp0 + BK, 0, 3);  // near-cache weight prefetch
  }

  // Spill gate tiles (D layout: v8f element r -> row 8*half+r, col l16)
#pragma unroll
  for (int mi = 0; mi < 2; ++mi)
#pragma unroll
    for (int ni = 0; ni < 2; ++ni)
#pragma unroll
      for (int r = 0; r < 8; ++r)
        sG[g][mh * 32 + mi * 16 + half * 8 + r][ni * 16 + l16] = acc[mi][ni].f[r];
  __syncthreads();

  // Fused elementwise epilogue: 2048 elems / 256 threads
  const float s0 = (phase == 1) ? scale[0] : 0.0f;
#pragma unroll
  for (int it = 0; it < 8; ++it) {
    const int e = tid + 256 * it;
    const int m = e >> 5;
    const int c = e & 31;
    const int i = mbase + m;
    const int j = nbase + c;      // 0..511 within gate block

    const float g0 = sG[0][m][c] + bias[j];
    const float g1 = sG[1][m][c] + bias[j + NG];
    const float g2 = sG[2][m][c] + bias[j + 2 * NG];
    const float g3 = sG[3][m][c] + bias[j + 3 * NG];

    const float forget = sigmoidf_(g0 + 1.0f);
    float ns = tanhf(g1);
    const float og = sigmoidf_(g3);

    if (hasP) {
      float wc;
      if (hasD) {
        const float alpha = sigmoidf_(g2);
        const float pc = prevC[(size_t)i * 512 + j];
        const float dc = delC[(size_t)i * 512 + j];
        wc = alpha * pc + (1.0f - alpha) * dc;
      } else {
        wc = prevC[(size_t)i * 512 + j];
      }
      ns = forget * wc + (1.0f - forget) * ns;
    }
    const float whole = og * ns;

    outNS[(size_t)i * 512 + j] = ns;
    if (j < 256) {
      if (phase == 1) {
        const float v = rawInput[(size_t)i * 256 + j] * __expf(whole * s0);
        outLowB[(size_t)i * DXH_K + j] = bf16_1(v);   // input2 -> xh2 (bf16)
      } else {
        outLowF[(size_t)i * 256 + j] = whole;         // output_t
      }
    } else {
      outHigh[(size_t)i * 256 + (j - 256)] = whole;   // hState / hState2
    }
  }
}

// ---------------- host-side launcher ----------------

extern "C" void kernel_launch(void* const* d_in, const int* in_sizes, int n_in,
                              void* d_out, int out_size, void* d_ws, size_t ws_size,
                              hipStream_t stream) {
  (void)in_sizes; (void)n_in; (void)out_size; (void)ws_size;

  const float* input_t = (const float*)d_in[0];
  const float* prevH   = (const float*)d_in[1];
  const float* delH    = (const float*)d_in[2];
  const float* prevC   = (const float*)d_in[3];
  const float* delC    = (const float*)d_in[4];
  const float* prevH2  = (const float*)d_in[5];
  const float* delH2   = (const float*)d_in[6];
  const float* prevC2  = (const float*)d_in[7];
  const float* delC2   = (const float*)d_in[8];
  const float* W1      = (const float*)d_in[9];
  const float* b1      = (const float*)d_in[10];
  const float* W2      = (const float*)d_in[11];
  const float* b2      = (const float*)d_in[12];
  const float* scale   = (const float*)d_in[13];
  const int*   hasD    = (const int*)d_in[14];
  const int*   hasP    = (const int*)d_in[15];

  float* out = (float*)d_out;
  float* out_output    = out;                           // (B,256) output_t
  float* out_hState    = out + (size_t)B_ROWS * 256;    // (B,256)
  float* out_newState  = out + (size_t)B_ROWS * 512;    // (B,512)
  float* out_hState2   = out + (size_t)B_ROWS * 1024;   // (B,256)
  float* out_newState2 = out + (size_t)B_ROWS * 1280;   // (B,512)

  // Workspace layout (bf16): W1b | W2b | xh1 | xh2  (~54 MB)
  unsigned short* W1b  = (unsigned short*)d_ws;
  unsigned short* W2b  = W1b + (size_t)4 * NG * DXH_K;
  unsigned short* xh1b = W2b + (size_t)4 * NG * DXH_K;
  unsigned short* xh2b = xh1b + (size_t)B_ROWS * DXH_K;

  const int wN8 = (4 * NG * DXH_K) / 8;                 // 196608
  hipLaunchKernelGGL(convert_f32_bf16, dim3((wN8 + 255) / 256), dim3(256), 0, stream, W1, W1b, wN8);
  hipLaunchKernelGGL(convert_f32_bf16, dim3((wN8 + 255) / 256), dim3(256), 0, stream, W2, W2b, wN8);

  {
    const int segs1 = DXH_K / 8;                        // 96
    const long long t1 = (long long)B_ROWS * segs1;
    hipLaunchKernelGGL(build_xh_bf16, dim3((unsigned)((t1 + 255) / 256)), dim3(256), 0, stream,
                       input_t, prevH, delH, xh1b, 0, hasD, hasP);
    const int segs2 = (DXH_K - 256) / 8;                // 64
    const long long t2 = (long long)B_ROWS * segs2;
    hipLaunchKernelGGL(build_xh_bf16, dim3((unsigned)((t2 + 255) / 256)), dim3(256), 0, stream,
                       input_t, prevH2, delH2, xh2b, 256, hasD, hasP);
  }

  dim3 grid(B_ROWS / BM, NG / BN);   // 256 x 16
  dim3 block(256);                   // 8 wave32s

  hipLaunchKernelGGL(s3cell_gate_gemm, grid, block, 0, stream,
      1, xh1b, W1b, b1, prevC, delC, input_t, scale,
      xh2b, (float*)nullptr, out_hState, out_newState, hasD, hasP);

  hipLaunchKernelGGL(s3cell_gate_gemm, grid, block, 0, stream,
      2, xh2b, W2b, b2, prevC2, delC2, (const float*)nullptr, (const float*)nullptr,
      (unsigned short*)nullptr, out_output, out_hState2, out_newState2, hasD, hasP);
}
